// BERTAttention_5961414607609
// MI455X (gfx1250) — compile-verified, matrix operand order checked by reading, not run on hardware
//
#include <hip/hip_runtime.h>
#include <math.h>

#define H     1024
#define NHEAD 16
#define HD    64
#define BATCH 8
#define SEQ   1024
#define MROWS (BATCH * SEQ)   // 8192

typedef __attribute__((ext_vector_type(16))) __bf16 v16bf;
typedef __attribute__((ext_vector_type(8)))  float  v8f;

// ---------------------------------------------------------------- helpers ---

static __device__ __forceinline__ unsigned short f2bf(float f) {
  union { float f; unsigned u; } x; x.f = f;
  unsigned r = x.u + 0x7FFFu + ((x.u >> 16) & 1u);   // round-to-nearest-even
  return (unsigned short)(r >> 16);
}

static __device__ __forceinline__ v8f wmma_bf16(v16bf a, v16bf b, v8f c) {
  // D = A(16x32 bf16) * B(32x16 bf16) + C(16x16 f32)
  return __builtin_amdgcn_wmma_f32_16x16x32_bf16(false, a, false, b,
                                                 (short)0, c, false, false);
}

// Load one 16x32 bf16 WMMA A/B fragment from a row-major bf16 matrix whose
// 32 K-elements per row are contiguous. Per the CDNA5 16-bit layout:
//   lanes 0-15 hold K = {kb+0..7, kb+16..23}, lanes 16-31 hold K = {+8, +24}.
static __device__ __forceinline__ v16bf load_frag_g(const unsigned short* base,
                                                    int row, int row_stride,
                                                    int kb, int lane) {
  const unsigned short* p =
      base + (size_t)row * row_stride + kb + ((lane >> 4) << 3);
  v16bf f;
  reinterpret_cast<uint4*>(&f)[0] = *reinterpret_cast<const uint4*>(p);
  reinterpret_cast<uint4*>(&f)[1] = *reinterpret_cast<const uint4*>(p + 16);
  return f;
}

// -------------------------------------------------------------- fp32->bf16 --

__global__ void cvt_kernel(const float* __restrict__ in,
                           unsigned short* __restrict__ out, int n) {
  int i = blockIdx.x * blockDim.x + threadIdx.x;
  int stride = gridDim.x * blockDim.x;
  for (; i < n; i += stride) out[i] = f2bf(in[i]);
}

// --------------------------------------------------------------- QKV GEMM ---
// grid = (MROWS/16, 3), block = 512 (16 waves). Wave w: rows m0..m0+15,
// cols [w*64, w*64+64). Rolled K-loop with ping-pong double buffering:
// the loop-carried buffers keep ~10 loads in flight behind each WMMA.
// Q,K stored [B,NH,S,HD]; V stored [B,NH,HD,S].

__global__ __launch_bounds__(512) void qkv_kernel(
    const unsigned short* __restrict__ Xbf,
    const unsigned short* __restrict__ Wqkv,
    const float* __restrict__ bq, const float* __restrict__ bk,
    const float* __restrict__ bv,
    unsigned short* __restrict__ Qbf, unsigned short* __restrict__ Kbf,
    unsigned short* __restrict__ Vt) {
  int mat = blockIdx.y;                         // 0=Q, 1=K, 2=V
  const unsigned short* W = Wqkv + (size_t)mat * H * H;
  const float* bias = (mat == 0) ? bq : (mat == 1) ? bk : bv;

  int m0 = blockIdx.x << 4;
  int tid = threadIdx.x;
  int w = tid >> 5, lane = tid & 31, l15 = lane & 15;
  int n0 = w << 6;                              // 64 columns per wave

  v8f acc[4] = {};

  // buffer set A (kb) and set B (kb+32), alternated; no rotation copies
  v16bf aA = load_frag_g(Xbf, m0 + l15, H, 0, lane);
  v16bf bA[4];
#pragma unroll
  for (int t = 0; t < 4; t++)
    bA[t] = load_frag_g(W, n0 + t * 16 + l15, H, 0, lane);

  for (int kb = 0; kb < H; kb += 64) {
    v16bf aB = load_frag_g(Xbf, m0 + l15, H, kb + 32, lane);
    v16bf bB[4];
#pragma unroll
    for (int t = 0; t < 4; t++)
      bB[t] = load_frag_g(W, n0 + t * 16 + l15, H, kb + 32, lane);

#pragma unroll
    for (int t = 0; t < 4; t++) acc[t] = wmma_bf16(aA, bA[t], acc[t]);

    int k2 = (kb + 64) & (H - 1);               // wraps to 0 on last iter
    aA = load_frag_g(Xbf, m0 + l15, H, k2, lane);
#pragma unroll
    for (int t = 0; t < 4; t++)
      bA[t] = load_frag_g(W, n0 + t * 16 + l15, H, k2, lane);

#pragma unroll
    for (int t = 0; t < 4; t++) acc[t] = wmma_bf16(aB, bB[t], acc[t]);
  }

  int rw = (lane >> 4) << 3;
#pragma unroll
  for (int t = 0; t < 4; t++) {
    int n = n0 + t * 16 + l15;
    int hh = n >> 6, d = n & 63;
    float bn = bias[n];
#pragma unroll
    for (int v = 0; v < 8; v++) {
      int row = m0 + v + rw;
      int bb = row >> 10, s = row & (SEQ - 1);
      unsigned short val = f2bf(acc[t][v] + bn);
      if (mat == 2)
        Vt[(((size_t)bb * NHEAD + hh) * HD + d) * SEQ + s] = val;    // V^T
      else if (mat == 0)
        Qbf[(((size_t)bb * NHEAD + hh) * SEQ + s) * HD + d] = val;
      else
        Kbf[(((size_t)bb * NHEAD + hh) * SEQ + s) * HD + d] = val;
    }
  }
}

// ------------------------------------------------------- flash attention ----
// One wave = one (b, h, 16-query tile). Online softmax over 32-key steps.
// V frags issued at top of iteration, next K frags issued before the softmax
// math; unroll-by-2 lets the allocator coalesce the K ping-pong (no movs).
// grid = B*NH*(SEQ/16)/8 = 1024 blocks of 256 threads.

__global__ __launch_bounds__(256) void attn_kernel(
    const unsigned short* __restrict__ Qbf,
    const unsigned short* __restrict__ Kbf,
    const unsigned short* __restrict__ Vt,
    const float* __restrict__ mask,
    unsigned short* __restrict__ Ctx) {
  __shared__ uint4 ldsp[8][64];                 // 1KB per wave: 16x32 bf16 P

  int tid = threadIdx.x;
  int w = tid >> 5, lane = tid & 31, l15 = lane & 15;
  int gid = blockIdx.x * 8 + w;
  int qt = gid & 63;
  int bh = gid >> 6;
  int b = bh >> 4, h = bh & 15;
  int q0 = qt << 4;

  const unsigned short* Qh = Qbf + (size_t)bh * SEQ * HD;
  const unsigned short* Kh = Kbf + (size_t)bh * SEQ * HD;
  const unsigned short* Vh = Vt  + (size_t)bh * HD * SEQ;
  const float* mrow = mask + b * SEQ;

  v16bf qa0 = load_frag_g(Qh, q0 + l15, HD, 0,  lane);
  v16bf qa1 = load_frag_g(Qh, q0 + l15, HD, 32, lane);

  v8f acc0 = {}, acc1 = {}, acc2 = {}, acc3 = {};
  float rmax[8], rsum[8];
#pragma unroll
  for (int v = 0; v < 8; v++) { rmax[v] = -INFINITY; rsum[v] = 0.f; }

  unsigned short* lw = (unsigned short*)&ldsp[w][0];
  int rw = (lane >> 4) << 3;

  // prime K fragments for j = 0
  v16bf kc0 = load_frag_g(Kh, 0 + l15,  HD, 0,  lane);
  v16bf kc1 = load_frag_g(Kh, 0 + l15,  HD, 32, lane);
  v16bf kc2 = load_frag_g(Kh, 16 + l15, HD, 0,  lane);
  v16bf kc3 = load_frag_g(Kh, 16 + l15, HD, 32, lane);

#pragma unroll 2
  for (int j = 0; j < SEQ; j += 32) {
    // V fragments for this step: independent of P, issue early
    v16bf vb0 = load_frag_g(Vh, 0 * 16 + l15, SEQ, j, lane);
    v16bf vb1 = load_frag_g(Vh, 1 * 16 + l15, SEQ, j, lane);
    v16bf vb2 = load_frag_g(Vh, 2 * 16 + l15, SEQ, j, lane);
    v16bf vb3 = load_frag_g(Vh, 3 * 16 + l15, SEQ, j, lane);

    // K fragments for the next step (wraps on last iter; redundant but cheap)
    int jn = (j + 32) & (SEQ - 1);
    v16bf kn0 = load_frag_g(Kh, jn + l15,      HD, 0,  lane);
    v16bf kn1 = load_frag_g(Kh, jn + l15,      HD, 32, lane);
    v16bf kn2 = load_frag_g(Kh, jn + 16 + l15, HD, 0,  lane);
    v16bf kn3 = load_frag_g(Kh, jn + 16 + l15, HD, 32, lane);

    // scores: Q(16x64) @ K^T for two 16-key sub-tiles
    v8f s0 = {}, s1 = {};
    s0 = wmma_bf16(qa0, kc0, s0);
    s0 = wmma_bf16(qa1, kc1, s0);
    s1 = wmma_bf16(qa0, kc2, s1);
    s1 = wmma_bf16(qa1, kc3, s1);

    float mk0 = mrow[j + l15];
    float mk1 = mrow[j + 16 + l15];

    float p0[8], p1[8];
#pragma unroll
    for (int v = 0; v < 8; v++) {
      float a = s0[v] * 0.125f + mk0;           // 1/sqrt(64) scale + mask
      float c = s1[v] * 0.125f + mk1;
      float t = fmaxf(a, c);
#pragma unroll
      for (int o = 1; o < 16; o <<= 1) t = fmaxf(t, __shfl_xor(t, o, 32));
      float nm = fmaxf(rmax[v], t);
      float alpha = __expf(rmax[v] - nm);
      rmax[v] = nm;
      float e0 = __expf(a - nm);
      float e1 = __expf(c - nm);
      p0[v] = e0; p1[v] = e1;
      float rs = e0 + e1;
#pragma unroll
      for (int o = 1; o < 16; o <<= 1) rs += __shfl_xor(rs, o, 32);
      rsum[v] = rsum[v] * alpha + rs;
      acc0[v] *= alpha; acc1[v] *= alpha; acc2[v] *= alpha; acc3[v] *= alpha;
    }

    // transpose P (C/D layout) -> A layout via per-wave LDS tile
#pragma unroll
    for (int v = 0; v < 8; v++) {
      int row = v + rw;
      lw[row * 32 + l15]      = f2bf(p0[v]);
      lw[row * 32 + 16 + l15] = f2bf(p1[v]);
    }
    asm volatile("s_wait_dscnt 0" ::: "memory");   // in-wave LDS RAW
    v16bf pa;
    {
      int idx = l15 * 4 + (lane >> 4);
      reinterpret_cast<uint4*>(&pa)[0] = ldsp[w][idx];
      reinterpret_cast<uint4*>(&pa)[1] = ldsp[w][idx + 2];
    }

    // ctx += P(16x32) @ V(32x64); V^T rows are d, contiguous keys
    acc0 = wmma_bf16(pa, vb0, acc0);
    acc1 = wmma_bf16(pa, vb1, acc1);
    acc2 = wmma_bf16(pa, vb2, acc2);
    acc3 = wmma_bf16(pa, vb3, acc3);

    kc0 = kn0; kc1 = kn1; kc2 = kn2; kc3 = kn3;
  }

  // normalize and store merged-head ctx [B*S, H] in bf16
#pragma unroll
  for (int v = 0; v < 8; v++) {
    float inv = 1.f / rsum[v];
    int s = q0 + v + rw;
    size_t rbase = ((size_t)b * SEQ + s) * H + h * HD;
    Ctx[rbase + 0 * 16 + l15] = f2bf(acc0[v] * inv);
    Ctx[rbase + 1 * 16 + l15] = f2bf(acc1[v] * inv);
    Ctx[rbase + 2 * 16 + l15] = f2bf(acc2[v] * inv);
    Ctx[rbase + 3 * 16 + l15] = f2bf(acc3[v] * inv);
  }
}

// ------------------------------- output GEMM + bias + residual + LayerNorm --
// grid = MROWS/16 blocks of 512 (16 waves). Wave w: 16 rows x 64 cols;
// block owns the full 16x1024 strip for the fused LayerNorm.
// Rolled ping-pong K-loop like qkv_kernel.

__global__ __launch_bounds__(512) void out_ln_kernel(
    const unsigned short* __restrict__ Ctx,
    const unsigned short* __restrict__ Wo,
    const float* __restrict__ bo,
    const float* __restrict__ hidden,
    const float* __restrict__ ln_g, const float* __restrict__ ln_b,
    float* __restrict__ out) {
  __shared__ float tile[16 * H];                // 64 KB

  int m0 = blockIdx.x << 4;
  int tid = threadIdx.x;
  int w = tid >> 5, lane = tid & 31, l15 = lane & 15;
  int n0 = w << 6;

  v8f acc[4] = {};
  v16bf aA = load_frag_g(Ctx, m0 + l15, H, 0, lane);
  v16bf bA[4];
#pragma unroll
  for (int t = 0; t < 4; t++)
    bA[t] = load_frag_g(Wo, n0 + t * 16 + l15, H, 0, lane);

  for (int kb = 0; kb < H; kb += 64) {
    v16bf aB = load_frag_g(Ctx, m0 + l15, H, kb + 32, lane);
    v16bf bB[4];
#pragma unroll
    for (int t = 0; t < 4; t++)
      bB[t] = load_frag_g(Wo, n0 + t * 16 + l15, H, kb + 32, lane);

#pragma unroll
    for (int t = 0; t < 4; t++) acc[t] = wmma_bf16(aA, bA[t], acc[t]);

    int k2 = (kb + 64) & (H - 1);
    aA = load_frag_g(Ctx, m0 + l15, H, k2, lane);
#pragma unroll
    for (int t = 0; t < 4; t++)
      bA[t] = load_frag_g(Wo, n0 + t * 16 + l15, H, k2, lane);

#pragma unroll
    for (int t = 0; t < 4; t++) acc[t] = wmma_bf16(aB, bB[t], acc[t]);
  }

  int rw = (lane >> 4) << 3;
#pragma unroll
  for (int t = 0; t < 4; t++) {
    int n = n0 + t * 16 + l15;
    float bn = bo[n];
#pragma unroll
    for (int v = 0; v < 8; v++) {
      int m = v + rw;
      int row = m0 + m;
      tile[m * H + n] = acc[t][v] + bn + hidden[(size_t)row * H + n];
    }
  }
  __syncthreads();

  // LayerNorm: 32 threads per row, 32 elems per thread
  int row = tid >> 5;                           // 16 rows
  const float* tr = &tile[row * H + lane * 32];
  float s = 0.f, ss = 0.f;
#pragma unroll 8
  for (int i = 0; i < 32; i++) { float x = tr[i]; s += x; ss += x * x; }
#pragma unroll
  for (int o = 1; o < 32; o <<= 1) {
    s  += __shfl_xor(s,  o, 32);
    ss += __shfl_xor(ss, o, 32);
  }
  float mu   = s  * (1.f / H);
  float var  = ss * (1.f / H) - mu * mu;
  float rstd = rsqrtf(var + 1e-12f);
  size_t obase = (size_t)(m0 + row) * H + lane * 32;
  for (int i = 0; i < 32; i++) {
    int n = lane * 32 + i;
    out[obase + i] = (tr[i] - mu) * rstd * ln_g[n] + ln_b[n];
  }
}

// ------------------------------------------------------------------ launch --

extern "C" void kernel_launch(void* const* d_in, const int* in_sizes, int n_in,
                              void* d_out, int out_size, void* d_ws,
                              size_t ws_size, hipStream_t stream) {
  (void)in_sizes; (void)n_in; (void)out_size; (void)ws_size;

  const float* hidden = (const float*)d_in[0];
  const float* mask   = (const float*)d_in[1];
  const float* Wq     = (const float*)d_in[2];
  const float* bq     = (const float*)d_in[3];
  const float* Wk     = (const float*)d_in[4];
  const float* bk     = (const float*)d_in[5];
  const float* Wv     = (const float*)d_in[6];
  const float* bv     = (const float*)d_in[7];
  const float* Wo     = (const float*)d_in[8];
  const float* bo     = (const float*)d_in[9];
  const float* ln_g   = (const float*)d_in[10];
  const float* ln_b   = (const float*)d_in[11];

  char* ws = (char*)d_ws;
  size_t off = 0;
  auto alloc = [&](size_t bytes) {
    char* p = ws + off;
    off += (bytes + 255) & ~(size_t)255;
    return p;
  };
  unsigned short* Xbf  = (unsigned short*)alloc((size_t)MROWS * H * 2);
  unsigned short* Wqkv = (unsigned short*)alloc((size_t)3 * H * H * 2);
  unsigned short* Wobf = (unsigned short*)alloc((size_t)H * H * 2);
  unsigned short* Qbf  = (unsigned short*)alloc((size_t)MROWS * H * 2);
  unsigned short* Kbf  = (unsigned short*)alloc((size_t)MROWS * H * 2);
  unsigned short* Vt   = (unsigned short*)alloc((size_t)MROWS * H * 2);
  unsigned short* Ctx  = (unsigned short*)alloc((size_t)MROWS * H * 2);

  cvt_kernel<<<dim3(4096), dim3(256), 0, stream>>>(hidden, Xbf, MROWS * H);
  cvt_kernel<<<dim3(1024), dim3(256), 0, stream>>>(Wq, Wqkv + 0 * H * H, H * H);
  cvt_kernel<<<dim3(1024), dim3(256), 0, stream>>>(Wk, Wqkv + 1 * H * H, H * H);
  cvt_kernel<<<dim3(1024), dim3(256), 0, stream>>>(Wv, Wqkv + 2 * H * H, H * H);
  cvt_kernel<<<dim3(1024), dim3(256), 0, stream>>>(Wo, Wobf, H * H);

  qkv_kernel<<<dim3(MROWS / 16, 3), dim3(512), 0, stream>>>(
      Xbf, Wqkv, bq, bk, bv, Qbf, Kbf, Vt);

  attn_kernel<<<dim3(BATCH * NHEAD * (SEQ / 16) / 8), dim3(256), 0, stream>>>(
      Qbf, Kbf, Vt, mask, Ctx);

  out_ln_kernel<<<dim3(MROWS / 16), dim3(512), 0, stream>>>(
      Ctx, Wobf, bo, hidden, ln_g, ln_b, (float*)d_out);
}
